// SCLSTM_69209103007803
// MI455X (gfx1250) — compile-verified, matrix operand order checked by reading, not run on hardware
//
#include <hip/hip_runtime.h>
#include <hip/hip_bf16.h>

// ---------------------------------------------------------------------------
// SC-LSTM on MI455X (gfx1250, wave32, WMMA + async LDS copies).
//   (1) f32->f16 weight/embedding conversion kernels.
//   (2) Batched input projections with WMMA f16 GEMM (M=1760).
//   (3) 55 sequential step kernels for the recurrence (LDS-resident state).
//   (4) Output projection 1760x32000x512 via WMMA GEMM with double-buffered
//       GLOBAL_LOAD_ASYNC_TO_LDS_B128 staging (ASYNCcnt + barriers).
// ---------------------------------------------------------------------------

typedef __attribute__((ext_vector_type(16))) _Float16 v16h;
typedef __attribute__((ext_vector_type(8)))  float    v8f;

#define BB 32
#define TT 55
#define VV 32000
#define DW 256
#define DD 512
#define AA 200
#define AP 256          // A padded to 256 for GEMM N
#define GG 2048         // 4*D
#define MROWS (BB*TT)   // 1760

// ---------------------------------------------------------------------------
// f32 -> f16 conversion (grid-stride)
// ---------------------------------------------------------------------------
__global__ void cvt_f32_f16(const float* __restrict__ in, _Float16* __restrict__ out, size_t n) {
    size_t i = (size_t)blockIdx.x * blockDim.x + threadIdx.x;
    size_t stride = (size_t)gridDim.x * blockDim.x;
    for (; i < n; i += stride) out[i] = (_Float16)in[i];
}

// Pad W_w2h_r (200x256) into 256x256 f16, zero rows 200..255; pad bias too.
__global__ void pad_wr(const float* __restrict__ W, const float* __restrict__ b,
                       _Float16* __restrict__ Wp, float* __restrict__ bp) {
    int i = blockIdx.x * 256 + threadIdx.x;     // grid covers 256*256
    int n = i >> 8;
    int k = i & 255;
    Wp[i] = (n < AA) ? (_Float16)W[n * DW + k] : (_Float16)0.f;
    if (i < AP) bp[i] = (i < AA) ? b[i] : 0.f;
}

// Embedding gather -> f16 rows (row r = b*T + t)
__global__ void embed_gather(const int* __restrict__ tgt, const float* __restrict__ E,
                             _Float16* __restrict__ emb) {
    int r = blockIdx.x;                          // 0..1759
    int idx = tgt[r];
    const float* src = E + (size_t)idx * DW;
    emb[(size_t)r * DW + threadIdx.x] = (_Float16)src[threadIdx.x];   // 256 threads
}

// h0 = c0 = enc_output[:,0,:]; dt0 = act_vecs
__global__ void init_state(const float* __restrict__ enc, const float* __restrict__ act,
                           float* __restrict__ h, float* __restrict__ c, float* __restrict__ dt) {
    int i = blockIdx.x * 256 + threadIdx.x;
    if (i < BB * DD) { h[i] = enc[i]; c[i] = enc[i]; }
    if (i < BB * AA) dt[i] = act[i];
}

// ---------------------------------------------------------------------------
// WMMA GEMM: C[M,N] = A[M,K] * B[N,K]^T + bias[N]
// A, B f16 row-major; C f32 row-major (ldc == N). K % 32 == 0, N % 256 == 0.
// Block tile 64x256, 256 threads (8 waves, 2x4 grid of 32x64 wave tiles).
// Tiles staged via GLOBAL_LOAD_ASYNC_TO_LDS_B128 (ASYNCcnt), double-buffered.
// Fragment layouts per CDNA5 ISA 7.12.2.
// ---------------------------------------------------------------------------
#define LDSA 40           // 32 + 8 halfs padding
#define LDSB 40
#define ABUF (64  * LDSA) // halfs per A stage
#define BBUF (256 * LDSB) // halfs per B stage

union Frag16 { v16h v; float4 f4[2]; };

__device__ __forceinline__ void async_b128(const _Float16* gptr, const _Float16* lptr) {
    // LDS byte address = low 32 bits of the generic shared-memory address.
    uint32_t la = (uint32_t)(uintptr_t)lptr;
    uint64_t ga = (uint64_t)(uintptr_t)gptr;
    asm volatile("global_load_async_to_lds_b128 %0, %1, off"
                 :: "v"(la), "v"(ga) : "memory");
}

__global__ __launch_bounds__(256)
void gemm_f16_wmma(const _Float16* __restrict__ A, const _Float16* __restrict__ Bw,
                   const float* __restrict__ bias, float* __restrict__ C,
                   int M, int N, int K, int lda, int ldb) {
    __shared__ _Float16 sA[2 * ABUF];
    __shared__ _Float16 sB[2 * BBUF];

    const int tid     = threadIdx.x;
    const int wave    = tid >> 5;
    const int lane    = tid & 31;
    const int laneHi  = lane >> 4;     // selects K(+8) half of frags, M(+8) of C
    const int laneRow = lane & 15;
    const int m0 = blockIdx.x * 64;
    const int n0 = blockIdx.y * 256;
    const int wm = (wave >> 2) * 32;   // 0 / 32
    const int wn = (wave & 3)  * 64;   // 0 / 64 / 128 / 192

    v8f acc[2][4] = {};

    // producer indices: 64 A rows x 4 chunks = 256 loads; 256 B rows x 4 = 1024
    const int arow = tid >> 2;         // 0..63
    const int achk = (tid & 3) * 8;    // half offset within 32-wide K slice
    int ag = m0 + arow; if (ag >= M) ag = M - 1;     // clamp (stores are guarded)
    const size_t aoff = (size_t)ag * lda + achk;

    // stage issue: 1 A chunk + 4 B chunks per thread = 5 async b128 per stage
    auto issue = [&](int buf, int k0) {
        async_b128(&A[aoff + k0], &sA[buf * ABUF + arow * LDSA + achk]);
#pragma unroll
        for (int rep = 0; rep < 4; ++rep) {
            const int idx  = tid + rep * 256;
            const int brow = idx >> 2;               // 0..255
            const int bchk = (idx & 3) * 8;
            async_b128(&Bw[(size_t)(n0 + brow) * ldb + k0 + bchk],
                       &sB[buf * BBUF + brow * LDSB + bchk]);
        }
    };

    issue(0, 0);
    int cur = 0;
    for (int k0 = 0; k0 < K; k0 += 32) {
        if (k0 + 32 < K) {
            issue(cur ^ 1, k0 + 32);
            // 5 newest (next stage) may remain outstanding; current stage done.
            asm volatile("s_wait_asynccnt 0x5" ::: "memory");
        } else {
            asm volatile("s_wait_asynccnt 0x0" ::: "memory");
        }
        __syncthreads();                             // all waves see current tile

        Frag16 a[2];
#pragma unroll
        for (int i = 0; i < 2; ++i) {
            const _Float16* p = &sA[cur * ABUF + (wm + i * 16 + laneRow) * LDSA + laneHi * 8];
            a[i].f4[0] = *(const float4*)(p);        // K +0..7   (+8 for hi lanes)
            a[i].f4[1] = *(const float4*)(p + 16);   // K +16..23 (+8 for hi lanes)
        }
#pragma unroll
        for (int j = 0; j < 4; ++j) {
            Frag16 b;
            const _Float16* q = &sB[cur * BBUF + (wn + j * 16 + laneRow) * LDSB + laneHi * 8];
            b.f4[0] = *(const float4*)(q);
            b.f4[1] = *(const float4*)(q + 16);
            acc[0][j] = __builtin_amdgcn_wmma_f32_16x16x32_f16(
                false, a[0].v, false, b.v, (short)0, acc[0][j], false, false);
            acc[1][j] = __builtin_amdgcn_wmma_f32_16x16x32_f16(
                false, a[1].v, false, b.v, (short)0, acc[1][j], false, false);
        }
        __syncthreads();                             // reads done before overwrite
        cur ^= 1;
    }

    // Store: C/D layout -> VGPR r: M = r + 8*laneHi, N = laneRow
#pragma unroll
    for (int i = 0; i < 2; ++i) {
        const int mbase = m0 + wm + i * 16 + laneHi * 8;
#pragma unroll
        for (int j = 0; j < 4; ++j) {
            const int n  = n0 + wn + j * 16 + laneRow;
            const float bb = bias ? bias[n] : 0.f;
#pragma unroll
            for (int r = 0; r < 8; ++r) {
                const int m = mbase + r;
                if (m < M) C[(size_t)m * N + n] = acc[i][j][r] + bb;
            }
        }
    }
}

// ---------------------------------------------------------------------------
// One LSTM step: one workgroup (256 threads) per batch row b.
// ---------------------------------------------------------------------------
__device__ __forceinline__ float sigmoidf_(float x) { return 1.f / (1.f + __expf(-x)); }

__global__ __launch_bounds__(256)
void lstm_step(int t,
               const float* __restrict__ w4_all,  // [MROWS, GG]
               const float* __restrict__ wr_all,  // [MROWS, AP]
               const float* __restrict__ W_h2h,   // [GG, DD]
               const float* __restrict__ b_h2h,   // [GG]
               const float* __restrict__ W_h2h_r, // [AA, DD]
               const float* __restrict__ b_h2h_r, // [AA]
               const float* __restrict__ W_dc,    // [DD, AA]
               float* __restrict__ h_state, float* __restrict__ c_state,
               float* __restrict__ dt_state,
               _Float16* __restrict__ Hf16) {     // [MROWS, DD]
    __shared__ float sh[DD];
    __shared__ float sg[GG];
    __shared__ float sdt[AA];
    const int b   = blockIdx.x;
    const int tid = threadIdx.x;
    const size_t row = (size_t)b * TT + t;

    for (int i = tid; i < DD; i += 256) sh[i] = h_state[b * DD + i];
    __syncthreads();

    // g = w4 + h @ W_h2h.T + b_h2h   (2048 dots of 512)
    for (int j = tid; j < GG; j += 256) {
        const float* w = W_h2h + (size_t)j * DD;
        float acc = w4_all[row * GG + j] + b_h2h[j];
#pragma unroll 4
        for (int k = 0; k < DD; ++k) acc += sh[k] * w[k];
        sg[j] = acc;
    }
    // gate_r & dt update (200 dots of 512)
    for (int a = tid; a < AA; a += 256) {
        const float* w = W_h2h_r + (size_t)a * DD;
        float acc = wr_all[row * AP + a] + b_h2h_r[a];
#pragma unroll 4
        for (int k = 0; k < DD; ++k) acc += sh[k] * w[k];
        const float r = sigmoidf_(acc);
        const float d = r * dt_state[b * AA + a];
        dt_state[b * AA + a] = d;
        sdt[a] = d;
    }
    __syncthreads();

    // c, h update (512 dots of 200 for dt @ W_dc.T)
    for (int j = tid; j < DD; j += 256) {
        const float* w = W_dc + (size_t)j * AA;
        float acc = 0.f;
#pragma unroll 4
        for (int k = 0; k < AA; ++k) acc += sdt[k] * w[k];
        const float gi = sigmoidf_(sg[j]);
        const float gf = sigmoidf_(sg[DD + j]);
        const float go = sigmoidf_(sg[2 * DD + j]);
        const float ch = tanhf(sg[3 * DD + j]);
        const float c  = gf * c_state[b * DD + j] + gi * ch + tanhf(acc);
        const float h  = go * tanhf(c);
        c_state[b * DD + j] = c;
        h_state[b * DD + j] = h;
        Hf16[row * DD + j]  = (_Float16)h;
    }
}

// ---------------------------------------------------------------------------
// Host-side orchestration
// ---------------------------------------------------------------------------
extern "C" void kernel_launch(void* const* d_in, const int* in_sizes, int n_in,
                              void* d_out, int out_size, void* d_ws, size_t ws_size,
                              hipStream_t stream) {
    (void)in_sizes; (void)n_in; (void)out_size; (void)ws_size;
    const int*   tgt_seq  = (const int*)  d_in[0];
    const float* enc_out  = (const float*)d_in[1];
    const float* act_vecs = (const float*)d_in[2];
    const float* E        = (const float*)d_in[3];
    const float* W_w2h    = (const float*)d_in[4];
    const float* b_w2h    = (const float*)d_in[5];
    const float* W_h2h    = (const float*)d_in[6];
    const float* b_h2h    = (const float*)d_in[7];
    const float* W_w2h_r  = (const float*)d_in[8];
    const float* b_w2h_r  = (const float*)d_in[9];
    const float* W_h2h_r  = (const float*)d_in[10];
    const float* b_h2h_r  = (const float*)d_in[11];
    const float* W_dc     = (const float*)d_in[12];
    const float* W_out    = (const float*)d_in[13];
    const float* b_out    = (const float*)d_in[14];
    float* out = (float*)d_out;

    // workspace carve-up (256B aligned)
    char* p = (char*)d_ws;
    auto carve = [&](size_t bytes) { void* r = (void*)p; p += (bytes + 255) & ~(size_t)255; return r; };
    _Float16* Wout16 = (_Float16*)carve((size_t)VV * DD * 2);
    _Float16* Ww2h16 = (_Float16*)carve((size_t)GG * DW * 2);
    _Float16* Wr16   = (_Float16*)carve((size_t)AP * DW * 2);
    float*    br_pad = (float*)   carve((size_t)AP * 4);
    _Float16* emb16  = (_Float16*)carve((size_t)MROWS * DW * 2);
    float*    w4_all = (float*)   carve((size_t)MROWS * GG * 4);
    float*    wr_all = (float*)   carve((size_t)MROWS * AP * 4);
    _Float16* Hf16   = (_Float16*)carve((size_t)MROWS * DD * 2);
    float*    h_st   = (float*)   carve((size_t)BB * DD * 4);
    float*    c_st   = (float*)   carve((size_t)BB * DD * 4);
    float*    dt_st  = (float*)   carve((size_t)BB * AA * 4);

    // (1) conversions / init
    cvt_f32_f16<<<4096, 256, 0, stream>>>(W_out, Wout16, (size_t)VV * DD);
    cvt_f32_f16<<<2048, 256, 0, stream>>>(W_w2h, Ww2h16, (size_t)GG * DW);
    pad_wr<<<AP * DW / 256, 256, 0, stream>>>(W_w2h_r, b_w2h_r, Wr16, br_pad);
    embed_gather<<<MROWS, DW, 0, stream>>>(tgt_seq, E, emb16);
    init_state<<<(BB * DD + 255) / 256, 256, 0, stream>>>(enc_out, act_vecs, h_st, c_st, dt_st);

    // (2) batched input projections (WMMA)
    {
        dim3 g((MROWS + 63) / 64, GG / 256);
        gemm_f16_wmma<<<g, 256, 0, stream>>>(emb16, Ww2h16, b_w2h, w4_all,
                                             MROWS, GG, DW, DW, DW);
    }
    {
        dim3 g((MROWS + 63) / 64, AP / 256);
        gemm_f16_wmma<<<g, 256, 0, stream>>>(emb16, Wr16, br_pad, wr_all,
                                             MROWS, AP, DW, DW, DW);
    }

    // (3) recurrence: 55 stream-ordered step kernels (device-wide sync between)
    for (int t = 0; t < TT; ++t) {
        lstm_step<<<BB, 256, 0, stream>>>(t, w4_all, wr_all, W_h2h, b_h2h,
                                          W_h2h_r, b_h2h_r, W_dc,
                                          h_st, c_st, dt_st, Hf16);
    }

    // (4) output projection: 1760 x 32000 x 512 (WMMA, the 57.7 GFLOP bulk)
    {
        dim3 g((MROWS + 63) / 64, VV / 256);
        gemm_f16_wmma<<<g, 256, 0, stream>>>(Hf16, Wout16, b_out, out,
                                             MROWS, VV, DD, DD, DD);
    }
}
